// PointNetFeaturePropagation_20633022889987
// MI455X (gfx1250) — compile-verified, matrix operand order checked by reading, not run on hardware
//
#include <hip/hip_runtime.h>
#include <hip/hip_bf16.h>
#include <stdint.h>

// ---- problem constants (from reference) ----
#define BB      16
#define NN      4096
#define MM      1024
#define C1_DIM  256
#define C2_DIM  128
#define HH      256
#define BNROWS  (BB * NN)        // 65536 points
#define K1_DIM  (C1_DIM + C2_DIM) // 384
#define K2_DIM  HH                // 256
#define BN_EPS  1e-5f

typedef __attribute__((ext_vector_type(16))) __bf16 v16bf_t;
typedef __attribute__((ext_vector_type(8)))  float  v8f_t;

__device__ __forceinline__ unsigned short f32_to_bf16(float f) {
  unsigned int x = __float_as_uint(f);
  unsigned int r = (x + 0x7FFFu + ((x >> 16) & 1u)) >> 16;  // RNE
  return (unsigned short)r;
}

// ---------------------------------------------------------------------------
// Convert fp32 weights to bf16
// ---------------------------------------------------------------------------
__global__ void convert_bf16_kernel(const float* __restrict__ src,
                                    unsigned short* __restrict__ dst, int n) {
  int i = blockIdx.x * blockDim.x + threadIdx.x;
  if (i < n) dst[i] = f32_to_bf16(src[i]);
}

// ---------------------------------------------------------------------------
// kNN(K=3) + inverse-distance interpolation + concat -> Xcat bf16 [BN, 384]
// One block = 256 dst points of one batch. src_xyz batch tile staged in LDS.
// ---------------------------------------------------------------------------
__global__ __launch_bounds__(256)
void interp_kernel(const float* __restrict__ src_xyz,
                   const float* __restrict__ src_feat,
                   const float* __restrict__ dst_xyz,
                   const float* __restrict__ dst_feat,
                   unsigned short* __restrict__ Xcat) {
  __shared__ float s_xyz[MM * 3];      // 12 KB
  __shared__ int   s_idx[256 * 3];
  __shared__ float s_w[256 * 3];

  const int b = blockIdx.y;
  const int nBase = blockIdx.x * 256;
  const int t = threadIdx.x;

  const float* sx = src_xyz + (size_t)b * MM * 3;
  for (int i = t; i < MM * 3; i += 256) s_xyz[i] = sx[i];
  __syncthreads();

  // ---- phase 1: each thread finds top-3 nearest src points for its dst ----
  const int n = nBase + t;
  const float px = dst_xyz[((size_t)b * NN + n) * 3 + 0];
  const float py = dst_xyz[((size_t)b * NN + n) * 3 + 1];
  const float pz = dst_xyz[((size_t)b * NN + n) * 3 + 2];
  float d0 = 3.4e38f, d1 = 3.4e38f, d2 = 3.4e38f;
  int   i0 = 0, i1 = 0, i2 = 0;
  for (int m = 0; m < MM; ++m) {
    float dx = px - s_xyz[m * 3 + 0];
    float dy = py - s_xyz[m * 3 + 1];
    float dz = pz - s_xyz[m * 3 + 2];
    float d  = dx * dx + dy * dy + dz * dz;
    if (d < d0)      { d2 = d1; i2 = i1; d1 = d0; i1 = i0; d0 = d; i0 = m; }
    else if (d < d1) { d2 = d1; i2 = i1; d1 = d;  i1 = m; }
    else if (d < d2) { d2 = d;  i2 = m; }
  }
  float r0 = 1.0f / (d0 + 1e-8f);
  float r1 = 1.0f / (d1 + 1e-8f);
  float r2 = 1.0f / (d2 + 1e-8f);
  float rs = 1.0f / (r0 + r1 + r2);
  s_idx[t * 3 + 0] = i0; s_idx[t * 3 + 1] = i1; s_idx[t * 3 + 2] = i2;
  s_w[t * 3 + 0] = r0 * rs; s_w[t * 3 + 1] = r1 * rs; s_w[t * 3 + 2] = r2 * rs;
  __syncthreads();

  // ---- phase 2: cooperative, channel-coalesced gather + concat ----
  const float* F = src_feat + (size_t)b * MM * C1_DIM;
  for (int p = 0; p < 256; ++p) {
    const size_t row = (size_t)b * NN + (nBase + p);
    const int j0 = s_idx[p * 3 + 0], j1 = s_idx[p * 3 + 1], j2 = s_idx[p * 3 + 2];
    const float w0 = s_w[p * 3 + 0], w1 = s_w[p * 3 + 1], w2 = s_w[p * 3 + 2];
    float v = w0 * F[(size_t)j0 * C1_DIM + t]
            + w1 * F[(size_t)j1 * C1_DIM + t]
            + w2 * F[(size_t)j2 * C1_DIM + t];
    Xcat[row * K1_DIM + C2_DIM + t] = f32_to_bf16(v);             // interp cols
    if (t < C2_DIM)
      Xcat[row * K1_DIM + t] = f32_to_bf16(dst_feat[row * C2_DIM + t]); // dst cols
  }
}

// ---------------------------------------------------------------------------
// GEMM: Y[BN, 256] = X[BN, KDIM](bf16) * W[256, KDIM]^T(bf16) + bias (f32 acc)
// One wave -> 16x64 output strip (4 accumulators), v_wmma_f32_16x16x32_bf16.
// Fragment addressing follows the CDNA5 16-bit A / f32 C VGPR layouts.
// ---------------------------------------------------------------------------
template<int KDIM>
__global__ __launch_bounds__(256)
void gemm_bias_kernel(const unsigned short* __restrict__ X,
                      const unsigned short* __restrict__ W,
                      const float* __restrict__ bias,
                      float* __restrict__ Y) {
  const int lane = threadIdx.x & 31;
  const int wave = threadIdx.x >> 5;
  const int rowTile = blockIdx.x * 2 + (wave >> 2);  // 16-row tiles
  const int nBase   = (wave & 3) * 64;
  const int m    = lane & 15;
  const int half = lane >> 4;
  const size_t rowBase = (size_t)rowTile * 16;

  union AB { v16bf_t v; unsigned int u[8]; };

  // per-lane K element offsets within one 32-wide K tile (bf16 pairs)
  int koff[8];
#pragma unroll
  for (int v = 0; v < 8; ++v)
    koff[v] = (v < 4) ? (2 * v + 8 * half) : (16 + 2 * (v - 4) + 8 * half);

  v8f_t acc0 = {}, acc1 = {}, acc2 = {}, acc3 = {};
  const unsigned short* Arow = X + (rowBase + m) * KDIM;
  const unsigned short* Wr0 = W + (size_t)(nBase +  0 + m) * KDIM;
  const unsigned short* Wr1 = W + (size_t)(nBase + 16 + m) * KDIM;
  const unsigned short* Wr2 = W + (size_t)(nBase + 32 + m) * KDIM;
  const unsigned short* Wr3 = W + (size_t)(nBase + 48 + m) * KDIM;

#pragma unroll 2
  for (int kt = 0; kt < KDIM; kt += 32) {
    AB a, b0, b1, b2, b3;
#pragma unroll
    for (int v = 0; v < 8; ++v) {
      a.u[v]  = *(const unsigned int*)(Arow + kt + koff[v]);
      b0.u[v] = *(const unsigned int*)(Wr0 + kt + koff[v]);
      b1.u[v] = *(const unsigned int*)(Wr1 + kt + koff[v]);
      b2.u[v] = *(const unsigned int*)(Wr2 + kt + koff[v]);
      b3.u[v] = *(const unsigned int*)(Wr3 + kt + koff[v]);
    }
    acc0 = __builtin_amdgcn_wmma_f32_16x16x32_bf16(false, a.v, false, b0.v,
                                                   (short)0, acc0, false, false);
    acc1 = __builtin_amdgcn_wmma_f32_16x16x32_bf16(false, a.v, false, b1.v,
                                                   (short)0, acc1, false, false);
    acc2 = __builtin_amdgcn_wmma_f32_16x16x32_bf16(false, a.v, false, b2.v,
                                                   (short)0, acc2, false, false);
    acc3 = __builtin_amdgcn_wmma_f32_16x16x32_bf16(false, a.v, false, b3.v,
                                                   (short)0, acc3, false, false);
  }

  // epilogue: C/D layout -> row = v + 8*half, col = nBase + ntile*16 + m
  const int c0 = nBase + m;
  const float bb0 = bias[c0 +  0], bb1 = bias[c0 + 16];
  const float bb2 = bias[c0 + 32], bb3 = bias[c0 + 48];
#pragma unroll
  for (int v = 0; v < 8; ++v) {
    size_t r = rowBase + v + 8 * half;
    Y[r * HH + c0 +  0] = acc0[v] + bb0;
    Y[r * HH + c0 + 16] = acc1[v] + bb1;
    Y[r * HH + c0 + 32] = acc2[v] + bb2;
    Y[r * HH + c0 + 48] = acc3[v] + bb3;
  }
}

// ---------------------------------------------------------------------------
// Per-channel sum / sum^2 (BatchNorm training stats), f32 atomics
// ---------------------------------------------------------------------------
__global__ __launch_bounds__(256)
void stats_kernel(const float* __restrict__ Y, float* __restrict__ stats) {
  const int c = threadIdx.x;                 // channel
  const size_t r0 = (size_t)blockIdx.x * 256;
  float s = 0.f, s2 = 0.f;
  for (int r = 0; r < 256; ++r) {
    float v = Y[(r0 + r) * HH + c];          // coalesced across threads
    s += v; s2 += v * v;
  }
  atomicAdd(&stats[c], s);
  atomicAdd(&stats[HH + c], s2);
}

// ---------------------------------------------------------------------------
// BN + ReLU, writing bf16 (next-layer activations) or f32 (final output)
// ---------------------------------------------------------------------------
__global__ __launch_bounds__(256)
void bn_relu_bf16_kernel(const float* __restrict__ Y, const float* __restrict__ stats,
                         const float* __restrict__ gamma, const float* __restrict__ beta,
                         unsigned short* __restrict__ out) {
  size_t i = (size_t)blockIdx.x * 256 + threadIdx.x;
  int c = (int)(i & (HH - 1));
  float mean = stats[c] * (1.0f / BNROWS);
  float var  = stats[HH + c] * (1.0f / BNROWS) - mean * mean;
  float v = (Y[i] - mean) * rsqrtf(var + BN_EPS) * gamma[c] + beta[c];
  out[i] = f32_to_bf16(v > 0.f ? v : 0.f);
}

__global__ __launch_bounds__(256)
void bn_relu_f32_kernel(const float* __restrict__ Y, const float* __restrict__ stats,
                        const float* __restrict__ gamma, const float* __restrict__ beta,
                        float* __restrict__ out) {
  size_t i = (size_t)blockIdx.x * 256 + threadIdx.x;
  int c = (int)(i & (HH - 1));
  float mean = stats[c] * (1.0f / BNROWS);
  float var  = stats[HH + c] * (1.0f / BNROWS) - mean * mean;
  float v = (Y[i] - mean) * rsqrtf(var + BN_EPS) * gamma[c] + beta[c];
  out[i] = v > 0.f ? v : 0.f;
}

// ---------------------------------------------------------------------------
extern "C" void kernel_launch(void* const* d_in, const int* in_sizes, int n_in,
                              void* d_out, int out_size, void* d_ws, size_t ws_size,
                              hipStream_t stream) {
  const float* src_xyz  = (const float*)d_in[0];
  const float* src_feat = (const float*)d_in[1];
  const float* dst_xyz  = (const float*)d_in[2];
  const float* dst_feat = (const float*)d_in[3];
  const float* W1 = (const float*)d_in[4];
  const float* b1 = (const float*)d_in[5];
  const float* g1 = (const float*)d_in[6];
  const float* be1 = (const float*)d_in[7];
  const float* W2 = (const float*)d_in[8];
  const float* b2 = (const float*)d_in[9];
  const float* g2 = (const float*)d_in[10];
  const float* be2 = (const float*)d_in[11];

  uint8_t* ws = (uint8_t*)d_ws;
  size_t off = 0;
  unsigned short* Xcat = (unsigned short*)(ws + off); off += (size_t)BNROWS * K1_DIM * 2;
  unsigned short* W1b  = (unsigned short*)(ws + off); off += (size_t)HH * K1_DIM * 2;
  unsigned short* W2b  = (unsigned short*)(ws + off); off += (size_t)HH * K2_DIM * 2;
  float*          Ybuf = (float*)(ws + off);          off += (size_t)BNROWS * HH * 4;
  unsigned short* X2   = (unsigned short*)(ws + off); off += (size_t)BNROWS * HH * 2;
  float*          stat = (float*)(ws + off);          off += 2 * HH * 4;

  // 1) weights -> bf16
  convert_bf16_kernel<<<(HH * K1_DIM + 255) / 256, 256, 0, stream>>>(W1, W1b, HH * K1_DIM);
  convert_bf16_kernel<<<(HH * K2_DIM + 255) / 256, 256, 0, stream>>>(W2, W2b, HH * K2_DIM);

  // 2) kNN + interpolation + concat -> Xcat bf16 [BN, 384]
  interp_kernel<<<dim3(NN / 256, BB), 256, 0, stream>>>(src_xyz, src_feat,
                                                        dst_xyz, dst_feat, Xcat);

  // 3) layer 1 GEMM (WMMA bf16, f32 accum)
  gemm_bias_kernel<K1_DIM><<<BNROWS / 32, 256, 0, stream>>>(Xcat, W1b, b1, Ybuf);

  // 4) BN stats + BN/ReLU -> bf16 activations X2
  hipMemsetAsync(stat, 0, 2 * HH * sizeof(float), stream);
  stats_kernel<<<BNROWS / 256, 256, 0, stream>>>(Ybuf, stat);
  bn_relu_bf16_kernel<<<(size_t)BNROWS * HH / 256, 256, 0, stream>>>(Ybuf, stat, g1, be1, X2);

  // 5) layer 2 GEMM (reuse Ybuf)
  gemm_bias_kernel<K2_DIM><<<BNROWS / 32, 256, 0, stream>>>(X2, W2b, b2, Ybuf);

  // 6) BN stats + BN/ReLU -> fp32 output
  hipMemsetAsync(stat, 0, 2 * HH * sizeof(float), stream);
  stats_kernel<<<BNROWS / 256, 256, 0, stream>>>(Ybuf, stat);
  bn_relu_f32_kernel<<<(size_t)BNROWS * HH / 256, 256, 0, stream>>>(Ybuf, stat, g2, be2,
                                                                    (float*)d_out);
}
